// JambaMambaMixer_40561671144129
// MI455X (gfx1250) — matrix operand order repaired and checked by reading, not compiled, not run on hardware
//
#include <hip/hip_runtime.h>
#include <hip/hip_bf16.h>

typedef __attribute__((ext_vector_type(16))) __bf16 v16bf;
typedef __attribute__((ext_vector_type(8)))  float  v8f;

#define LTOK 2048
#define HDIM 2048
#define IDIM 4096
#define NST  16
#define KCONV 4
#define RRANK 128
#define SPN  160   // RRANK + 2*NST (logical)
#define SPAD 192   // padded to a multiple of 64 for the WMMA GEMM

// ---------------------------------------------------------------- helpers
__device__ __forceinline__ unsigned short f2bf(float f) {
    unsigned u = __builtin_bit_cast(unsigned, f);
    unsigned r = ((u & 0x7F800000u) == 0x7F800000u)
                     ? u
                     : (u + 0x7FFFu + ((u >> 16) & 1u));   // RNE
    return (unsigned short)(r >> 16);
}

__device__ __forceinline__ float silu_f(float x) {
    return x / (1.0f + __expf(-x));
}

// ---------------------------------------------------------------- fp32 -> bf16 (x4 vectorized)
__global__ void cvt_f32_bf16_x4(const float4* __restrict__ in,
                                ushort4* __restrict__ out, int n4) {
    int i = blockIdx.x * blockDim.x + threadIdx.x;
    if (i < n4) {
        float4 f = in[i];
        ushort4 o;
        o.x = f2bf(f.x); o.y = f2bf(f.y); o.z = f2bf(f.z); o.w = f2bf(f.w);
        out[i] = o;
    }
}

// ---------------------------------------------------------------- WMMA NT GEMM
// C[M,N] f32 = A[M,K] bf16 (row-major) * B[N,K] bf16 (row-major, transposed use)
// Wave tile 64x64 (4x4 WMMA accumulators), block = 8 waves -> 128x256 tile.
// Requires M % 128 == 0, N % 64 == 0, K % 32 == 0.
__global__ __launch_bounds__(256) void gemm_bf16_nt(
        const unsigned short* __restrict__ A,
        const unsigned short* __restrict__ B,
        float* __restrict__ C, int M, int N, int K) {
    const int lane = threadIdx.x & 31;
    const int wave = threadIdx.x >> 5;
    const int wm = wave >> 2;                  // 0..1
    const int wn = wave & 3;                   // 0..3
    const int m0 = blockIdx.y * 128 + wm * 64;
    const int n0 = blockIdx.x * 256 + wn * 64;
    if (m0 >= M || n0 >= N) return;            // wave-uniform skip

    const int kh  = lane >> 4;                 // 0/1
    const int l16 = lane & 15;

    union FragU { v16bf v; uint4 q[2]; };

    const v8f vzero = {0.f, 0.f, 0.f, 0.f, 0.f, 0.f, 0.f, 0.f};
    v8f acc[4][4];
#pragma unroll
    for (int i = 0; i < 4; ++i)
#pragma unroll
        for (int j = 0; j < 4; ++j) acc[i][j] = vzero;

    // Named per-fragment base pointers (keeps global address-space inference).
    // A frag: k = k0 + kh*16 (+8 for q[1]); B frag: k = k0 + kh*16 (+8 for q[1]).
    const unsigned short* pa0 = A + (size_t)(m0 +  0 + l16) * K + kh * 16;
    const unsigned short* pa1 = A + (size_t)(m0 + 16 + l16) * K + kh * 16;
    const unsigned short* pa2 = A + (size_t)(m0 + 32 + l16) * K + kh * 16;
    const unsigned short* pa3 = A + (size_t)(m0 + 48 + l16) * K + kh * 16;
    const unsigned short* pb0 = B + (size_t)(n0 +  0 + l16) * K + kh * 16;
    const unsigned short* pb1 = B + (size_t)(n0 + 16 + l16) * K + kh * 16;
    const unsigned short* pb2 = B + (size_t)(n0 + 32 + l16) * K + kh * 16;
    const unsigned short* pb3 = B + (size_t)(n0 + 48 + l16) * K + kh * 16;

#pragma unroll 1
    for (int k0 = 0; k0 < K; k0 += 32) {
        FragU a0, a1, a2, a3, b0, b1, b2, b3;
        a0.q[0] = *(const uint4*)(pa0); a0.q[1] = *(const uint4*)(pa0 + 8);
        a1.q[0] = *(const uint4*)(pa1); a1.q[1] = *(const uint4*)(pa1 + 8);
        a2.q[0] = *(const uint4*)(pa2); a2.q[1] = *(const uint4*)(pa2 + 8);
        a3.q[0] = *(const uint4*)(pa3); a3.q[1] = *(const uint4*)(pa3 + 8);
        b0.q[0] = *(const uint4*)(pb0); b0.q[1] = *(const uint4*)(pb0 + 8);
        b1.q[0] = *(const uint4*)(pb1); b1.q[1] = *(const uint4*)(pb1 + 8);
        b2.q[0] = *(const uint4*)(pb2); b2.q[1] = *(const uint4*)(pb2 + 8);
        b3.q[0] = *(const uint4*)(pb3); b3.q[1] = *(const uint4*)(pb3 + 8);

        if (k0 + 32 < K) {                     // warm next K-slab
            __builtin_prefetch(pa0 + 32, 0, 3);
            __builtin_prefetch(pb0 + 32, 0, 3);
        }

        acc[0][0] = __builtin_amdgcn_wmma_f32_16x16x32_bf16(false, a0.v, false, b0.v, (short)0, acc[0][0], false, false);
        acc[0][1] = __builtin_amdgcn_wmma_f32_16x16x32_bf16(false, a0.v, false, b1.v, (short)0, acc[0][1], false, false);
        acc[0][2] = __builtin_amdgcn_wmma_f32_16x16x32_bf16(false, a0.v, false, b2.v, (short)0, acc[0][2], false, false);
        acc[0][3] = __builtin_amdgcn_wmma_f32_16x16x32_bf16(false, a0.v, false, b3.v, (short)0, acc[0][3], false, false);
        acc[1][0] = __builtin_amdgcn_wmma_f32_16x16x32_bf16(false, a1.v, false, b0.v, (short)0, acc[1][0], false, false);
        acc[1][1] = __builtin_amdgcn_wmma_f32_16x16x32_bf16(false, a1.v, false, b1.v, (short)0, acc[1][1], false, false);
        acc[1][2] = __builtin_amdgcn_wmma_f32_16x16x32_bf16(false, a1.v, false, b2.v, (short)0, acc[1][2], false, false);
        acc[1][3] = __builtin_amdgcn_wmma_f32_16x16x32_bf16(false, a1.v, false, b3.v, (short)0, acc[1][3], false, false);
        acc[2][0] = __builtin_amdgcn_wmma_f32_16x16x32_bf16(false, a2.v, false, b0.v, (short)0, acc[2][0], false, false);
        acc[2][1] = __builtin_amdgcn_wmma_f32_16x16x32_bf16(false, a2.v, false, b1.v, (short)0, acc[2][1], false, false);
        acc[2][2] = __builtin_amdgcn_wmma_f32_16x16x32_bf16(false, a2.v, false, b2.v, (short)0, acc[2][2], false, false);
        acc[2][3] = __builtin_amdgcn_wmma_f32_16x16x32_bf16(false, a2.v, false, b3.v, (short)0, acc[2][3], false, false);
        acc[3][0] = __builtin_amdgcn_wmma_f32_16x16x32_bf16(false, a3.v, false, b0.v, (short)0, acc[3][0], false, false);
        acc[3][1] = __builtin_amdgcn_wmma_f32_16x16x32_bf16(false, a3.v, false, b1.v, (short)0, acc[3][1], false, false);
        acc[3][2] = __builtin_amdgcn_wmma_f32_16x16x32_bf16(false, a3.v, false, b2.v, (short)0, acc[3][2], false, false);
        acc[3][3] = __builtin_amdgcn_wmma_f32_16x16x32_bf16(false, a3.v, false, b3.v, (short)0, acc[3][3], false, false);

        pa0 += 32; pa1 += 32; pa2 += 32; pa3 += 32;
        pb0 += 32; pb1 += 32; pb2 += 32; pb3 += 32;
    }

#pragma unroll
    for (int i = 0; i < 4; ++i) {
        float* cp = C + (size_t)(m0 + 16 * i + kh * 8) * N + n0 + l16;
#pragma unroll
        for (int j = 0; j < 4; ++j)
#pragma unroll
            for (int r = 0; r < 8; ++r)
                cp[(size_t)r * N + 16 * j] = acc[i][j][r];
    }
}

// ---------------------------------------------------------------- conv1d + SiLU
__global__ void conv_silu(const float* __restrict__ proj,
                          const float* __restrict__ cw,
                          const float* __restrict__ cb,
                          float* __restrict__ xs,
                          unsigned short* __restrict__ xsb) {
    int idx = blockIdx.x * blockDim.x + threadIdx.x;
    if (idx >= LTOK * IDIM) return;
    int c = idx % IDIM;
    int l = idx / IDIM;
    float acc = cb[c];
#pragma unroll
    for (int i = 0; i < KCONV; ++i) {
        int t = l - (KCONV - 1) + i;
        if (t >= 0)
            acc += cw[c * KCONV + i] * proj[(size_t)t * (2 * IDIM) + c];
    }
    float s = silu_f(acc);
    xs[idx]  = s;
    xsb[idx] = f2bf(s);
}

// ---------------------------------------------------------------- RMSNorm split
// sp[l, 0:160] (ld = SPAD) -> dt_r (bf16), Bm, Cm (f32). One block per token.
__global__ __launch_bounds__(128) void rms_split(
        const float* __restrict__ sp,
        const float* __restrict__ dtln, const float* __restrict__ bln,
        const float* __restrict__ cln,
        unsigned short* __restrict__ dtrb,
        float* __restrict__ Bm, float* __restrict__ Cm) {
    __shared__ float red[128];
    const int l = blockIdx.x;
    const int j = threadIdx.x;
    const float* row = sp + (size_t)l * SPAD;

    float v = row[j];
    red[j] = v * v;
    __syncthreads();
    for (int s = 64; s > 0; s >>= 1) {
        if (j < s) red[j] += red[j + s];
        __syncthreads();
    }
    float inv = rsqrtf(red[0] / (float)RRANK + 1e-6f);
    dtrb[(size_t)l * RRANK + j] = f2bf(v * inv * dtln[j]);

    if (j < NST) {
        float ssb = 0.f, ssc = 0.f;
#pragma unroll
        for (int i = 0; i < NST; ++i) {
            float b = row[RRANK + i];        ssb += b * b;
            float c = row[RRANK + NST + i];  ssc += c * c;
        }
        Bm[(size_t)l * NST + j] = row[RRANK + j] * rsqrtf(ssb / NST + 1e-6f) * bln[j];
        Cm[(size_t)l * NST + j] = row[RRANK + NST + j] * rsqrtf(ssc / NST + 1e-6f) * cln[j];
    }
}

// ---------------------------------------------------------------- softplus(dt + b)  (in-place)
__global__ void dt_softplus(float* __restrict__ dt, const float* __restrict__ bias) {
    int idx = blockIdx.x * blockDim.x + threadIdx.x;
    if (idx >= LTOK * IDIM) return;
    int i = idx % IDIM;
    float x = dt[idx] + bias[i];
    dt[idx] = (x > 20.f) ? x : log1pf(__expf(x));
}

// ---------------------------------------------------------------- selective scan
__global__ __launch_bounds__(256) void selective_scan(
        const float* __restrict__ xs, const float* __restrict__ dt,
        const float* __restrict__ Bm, const float* __restrict__ Cm,
        const float* __restrict__ Amat, float* __restrict__ ys) {
    const int c = blockIdx.x * 256 + threadIdx.x;   // < IDIM by construction
    __shared__ float Bs[NST], Cs[NST];
    float st[NST], a[NST];
#pragma unroll
    for (int n = 0; n < NST; ++n) {
        st[n] = 0.f;
        a[n]  = Amat[(size_t)c * NST + n];
    }
    for (int t = 0; t < LTOK; ++t) {
        if (threadIdx.x < NST)
            Bs[threadIdx.x] = Bm[(size_t)t * NST + threadIdx.x];
        else if (threadIdx.x < 2 * NST)
            Cs[threadIdx.x - NST] = Cm[(size_t)t * NST + threadIdx.x - NST];
        __syncthreads();
        float dtv = dt[(size_t)t * IDIM + c];
        float dx  = dtv * xs[(size_t)t * IDIM + c];
        float y = 0.f;
#pragma unroll
        for (int n = 0; n < NST; ++n) {
            st[n] = st[n] * __expf(dtv * a[n]) + dx * Bs[n];
            y += st[n] * Cs[n];
        }
        ys[(size_t)t * IDIM + c] = y;
        __syncthreads();
    }
}

// ---------------------------------------------------------------- gate epilogue
__global__ void gate_epilogue(const float* __restrict__ ys,
                              const float* __restrict__ xs,
                              const float* __restrict__ proj,
                              const float* __restrict__ D,
                              unsigned short* __restrict__ yb) {
    int idx = blockIdx.x * blockDim.x + threadIdx.x;
    if (idx >= LTOK * IDIM) return;
    int i = idx % IDIM;
    int l = idx / IDIM;
    float g = proj[(size_t)l * (2 * IDIM) + IDIM + i];
    float y = (ys[idx] + xs[idx] * D[i]) * silu_f(g);
    yb[idx] = f2bf(y);
}

// ---------------------------------------------------------------- launch
extern "C" void kernel_launch(void* const* d_in, const int* in_sizes, int n_in,
                              void* d_out, int out_size, void* d_ws, size_t ws_size,
                              hipStream_t stream) {
    constexpr int L = LTOK, H = HDIM, I = IDIM, Ns = NST, R = RRANK;

    const float* hs    = (const float*)d_in[0];
    const float* w_in  = (const float*)d_in[1];
    const float* cw    = (const float*)d_in[2];
    const float* cb    = (const float*)d_in[3];
    const float* w_xp  = (const float*)d_in[4];
    const float* dtln  = (const float*)d_in[5];
    const float* bln   = (const float*)d_in[6];
    const float* cln   = (const float*)d_in[7];
    const float* w_dt  = (const float*)d_in[8];
    const float* b_dt  = (const float*)d_in[9];
    const float* Amat  = (const float*)d_in[10];
    const float* Dvec  = (const float*)d_in[11];
    const float* w_out = (const float*)d_in[12];
    float* out = (float*)d_out;

    char* ws = (char*)d_ws;
    size_t off = 0;
    auto alloc = [&](size_t bytes) -> char* {
        char* p = ws + off;
        off += (bytes + 255) & ~(size_t)255;
        return p;
    };

    unsigned short* hsb   = (unsigned short*)alloc((size_t)L * H * 2);
    unsigned short* w1b   = (unsigned short*)alloc((size_t)2 * I * H * 2);
    unsigned short* xpwb  = (unsigned short*)alloc((size_t)SPAD * I * 2);  // 160 rows valid, 192 alloc'd
    unsigned short* dtpwb = (unsigned short*)alloc((size_t)I * R * 2);
    unsigned short* opwb  = (unsigned short*)alloc((size_t)H * I * 2);
    float*          proj  = (float*)alloc((size_t)L * 2 * I * 4);
    float*          xs    = (float*)alloc((size_t)L * I * 4);
    unsigned short* xsb   = (unsigned short*)alloc((size_t)L * I * 2);
    float*          sp    = (float*)alloc((size_t)L * SPAD * 4);           // ld = 192
    unsigned short* dtrb  = (unsigned short*)alloc((size_t)L * R * 2);
    float*          Bm    = (float*)alloc((size_t)L * Ns * 4);
    float*          Cm    = (float*)alloc((size_t)L * Ns * 4);
    float*          dtb   = (float*)alloc((size_t)L * I * 4);
    float*          ys    = (float*)alloc((size_t)L * I * 4);
    unsigned short* yb    = (unsigned short*)alloc((size_t)L * I * 2);
    (void)ws_size; (void)n_in; (void)in_sizes; (void)out_size;

    auto cvt = [&](const float* src, unsigned short* dst, int n) {
        int n4 = n / 4;
        cvt_f32_bf16_x4<<<(n4 + 255) / 256, 256, 0, stream>>>(
            (const float4*)src, (ushort4*)dst, n4);
    };
    cvt(hs,    hsb,   L * H);
    cvt(w_in,  w1b,   2 * I * H);
    cvt(w_xp,  xpwb,  SPN * I);       // rows 160..191 of xpwb stay garbage (unread cols)
    cvt(w_dt,  dtpwb, I * R);
    cvt(w_out, opwb,  H * I);

    auto gemm = [&](const unsigned short* A, const unsigned short* B, float* C,
                    int M, int N, int K) {
        dim3 grid((N + 255) / 256, M / 128);
        gemm_bf16_nt<<<grid, 256, 0, stream>>>(A, B, C, M, N, K);
    };

    // 1) in_proj: proj[L, 2I]
    gemm(hsb, w1b, proj, L, 2 * I, H);

    // 2) causal depthwise conv + SiLU
    conv_silu<<<(L * I + 255) / 256, 256, 0, stream>>>(proj, cw, cb, xs, xsb);

    // 3) x_proj: sp[L, 192] (cols 160..191 are padding, never read)
    gemm(xsb, xpwb, sp, L, SPAD, I);

    // 4) per-branch RMSNorm
    rms_split<<<L, 128, 0, stream>>>(sp, dtln, bln, cln, dtrb, Bm, Cm);

    // 5) dt_proj + softplus
    gemm(dtrb, dtpwb, dtb, L, I, R);
    dt_softplus<<<(L * I + 255) / 256, 256, 0, stream>>>(dtb, b_dt);

    // 6) selective scan
    selective_scan<<<I / 256, 256, 0, stream>>>(xs, dtb, Bm, Cm, Amat, ys);

    // 7) gate + D skip, to bf16
    gate_epilogue<<<(L * I + 255) / 256, 256, 0, stream>>>(ys, xs, proj, Dvec, yb);

    // 8) out_proj -> d_out
    gemm(yb, opwb, out, L, H, I);
}
